// NDimMultiHeadAttention_82540681494879
// MI455X (gfx1250) — matrix operand order repaired and checked by reading, not compiled, not run on hardware
//
#include <hip/hip_runtime.h>

// ---------------------------------------------------------------------------
// CDNA5 (gfx1250) bf16 WMMA attention pipeline.
// wave32; v_wmma_f32_16x16x32_bf16 for all matmuls; f32 softmax/accumulation;
// TDM (tensor_load_to_lds) stages the GEMM weight panel into LDS.
// ---------------------------------------------------------------------------

typedef __attribute__((ext_vector_type(16))) __bf16 v16bf;
typedef __attribute__((ext_vector_type(8)))  float  v8f;
typedef __attribute__((ext_vector_type(4)))  unsigned int v4u;
typedef __attribute__((ext_vector_type(8)))  int    v8i;
typedef __attribute__((ext_vector_type(4)))  int    v4i;

#define DEVINL __device__ __forceinline__

DEVINL unsigned short f2bf(float f) {
    // round-to-nearest-even f32 -> bf16
    unsigned int u = __float_as_uint(f);
    u += 0x7FFFu + ((u >> 16) & 1u);
    return (unsigned short)(u >> 16);
}

union FragBF {
    v16bf v;
    unsigned int u[8];
};

DEVINL v8f wmma_bf16(const FragBF& a, const FragBF& b, v8f c) {
    return __builtin_amdgcn_wmma_f32_16x16x32_bf16(
        /*neg_a=*/false, a.v, /*neg_b=*/false, b.v,
        /*c_mod=*/(short)0, c, /*reuse_a=*/false, /*reuse_b=*/false);
}

// ---------------------------------------------------------------------------
// Kernel 0: elementwise f32 -> bf16
// ---------------------------------------------------------------------------
__global__ void cvt_f32_to_bf16(const float* __restrict__ x,
                                unsigned short* __restrict__ y, int n) {
    int i = blockIdx.x * blockDim.x + threadIdx.x;
    int stride = gridDim.x * blockDim.x;
    for (; i < n; i += stride) y[i] = f2bf(x[i]);
}

// ---------------------------------------------------------------------------
// Kernel 1: bf16 GEMM  C = A(MxK) * B(NxK)^T   (i.e. x @ W.T)
//   - 128 threads = 4 waves; each wave computes a 32x32 tile (2x2 WMMA tiles)
//   - the 64-row B panel ([n0..n0+63] x K bf16, 64KB for K=512) is DMA'd into
//     LDS once per block by the Tensor Data Mover; tracked with TENSORcnt.
//   - M % 64 == 0, N % 64 == 0, K % 32 == 0, K <= 512 (K*64*2 <= 64KB LDS)
//   - Cf != nullptr : f32 output [M,N] (+ optional bias[N])
//   - else transC   : bf16 output stored transposed [N,M]
//   - else          : bf16 output [M,N]
// ---------------------------------------------------------------------------
__global__ void __launch_bounds__(128) gemm_bf16_wmma(
    const unsigned short* __restrict__ A,
    const unsigned short* __restrict__ B,
    unsigned short* __restrict__ Cb,
    float* __restrict__ Cf,
    const float* __restrict__ bias,
    int M, int N, int K, int transC) {
    const int lane = threadIdx.x & 31;
    const int wid  = threadIdx.x >> 5;
    const int lhi  = lane >> 4;     // which 16-lane half
    const int ll   = lane & 15;
    const int m0 = blockIdx.x * 64 + (wid >> 1) * 32;
    const int nloc0 = (wid & 1) * 32;            // n offset inside the LDS panel
    const int n0 = blockIdx.y * 64 + nloc0;      // global n offset

    __shared__ unsigned short Bs[64 * 512];      // B panel: [64 rows][K<=512] bf16

    // ---- TDM: DMA the B panel (64 x K bf16) into LDS (wave 0 only) ----
    if (wid == 0) {
        const unsigned long long ga =
            (unsigned long long)(uintptr_t)(B + (size_t)blockIdx.y * 64 * K);
        const unsigned ldsOff = (unsigned)(uintptr_t)(&Bs[0]);  // addr[31:0] = LDS byte offset
        // D# group 0: count=1 | lds_addr | global_addr[56:0] | type=2 (bits 127:126)
        v4u g0;
        g0[0] = 1u;                                   // count = 1 valid descriptor
        g0[1] = ldsOff;                               // LDS byte address
        g0[2] = (unsigned)(ga & 0xFFFFFFFFu);         // global_addr[31:0]
        g0[3] = (unsigned)((ga >> 32) & 0x01FFFFFFu)  // global_addr[56:32]
                | 0x80000000u;                        // type = 2 ("image")
        // D# group 1: data_size=2B, dims/strides in elements
        const unsigned TD0 = (unsigned)K;   // tensor dim0 (contiguous)
        const unsigned TD1 = (unsigned)N;   // tensor dim1 (rows)
        const unsigned TL0 = (unsigned)K;   // tile dim0 = whole rows
        const unsigned TL1 = 64u;           // tile dim1 = 64 rows
        v8i g1;
        g1[0] = 0x00010000;                                   // data_size code 1 = 2 bytes
        g1[1] = (int)((TD0 & 0xFFFFu) << 16);                 // tensor_dim0[15:0]
        g1[2] = (int)((TD0 >> 16) | ((TD1 & 0xFFFFu) << 16)); // dim0[31:16] | dim1[15:0]
        g1[3] = (int)((TD1 >> 16) | (TL0 << 16));             // dim1[31:16] | tile_dim0
        g1[4] = (int)TL1;                                     // tile_dim1 | tile_dim2=0
        g1[5] = (int)TD0;                                     // tensor_dim0_stride[31:0]
        g1[6] = 0;                                            // stride0[47:32] | stride1[15:0]
        g1[7] = 0;                                            // stride1[47:16]
        v4i g2 = {0, 0, 0, 0};                                // 2D tensor: dims 2/3 unused
        v4i g3 = {0, 0, 0, 0};
        v8i g4 = {0, 0, 0, 0, 0, 0, 0, 0};                    // extra group (6-arg form)
        __builtin_amdgcn_tensor_load_to_lds(g0, g1, g2, g3, g4, 0);
        __builtin_amdgcn_s_wait_tensorcnt(0);
    }
    __syncthreads();

    v8f acc[2][2];
#pragma unroll
    for (int mi = 0; mi < 2; ++mi)
#pragma unroll
        for (int ni = 0; ni < 2; ++ni)
#pragma unroll
            for (int j = 0; j < 8; ++j) acc[mi][ni][j] = 0.f;

    for (int k0 = 0; k0 < K; k0 += 32) {
        FragBF a[2], bb[2];
#pragma unroll
        for (int mi = 0; mi < 2; ++mi) {
            const unsigned short* Ar = A + (size_t)(m0 + mi * 16 + ll) * K + k0;
            __builtin_prefetch(Ar + 32, 0, 1);   // next k panel -> global_prefetch_b8
#pragma unroll
            for (int j = 0; j < 8; ++j) {
                // A in contiguous k order (same as B): VGPR j holds K = 2j (+16 for hi lanes)
                const int kA = 2 * j + lhi * 16;
                a[mi].u[j] = *reinterpret_cast<const unsigned int*>(Ar + kA);
            }
        }
#pragma unroll
        for (int ni = 0; ni < 2; ++ni) {
            // B layout: lane = column, VGPR j holds K = 2j (+16 for hi lanes); from LDS panel
            const unsigned short* Br = &Bs[(nloc0 + ni * 16 + ll) * K + k0 + lhi * 16];
#pragma unroll
            for (int j = 0; j < 8; ++j)
                bb[ni].u[j] = *reinterpret_cast<const unsigned int*>(Br + 2 * j);
        }
#pragma unroll
        for (int mi = 0; mi < 2; ++mi)
#pragma unroll
            for (int ni = 0; ni < 2; ++ni)
                acc[mi][ni] = wmma_bf16(a[mi], bb[ni], acc[mi][ni]);
    }

#pragma unroll
    for (int mi = 0; mi < 2; ++mi)
#pragma unroll
        for (int ni = 0; ni < 2; ++ni)
#pragma unroll
            for (int j = 0; j < 8; ++j) {
                const int m = m0 + mi * 16 + j + lhi * 8;   // C layout: row = j (+8 hi lanes)
                const int n = n0 + ni * 16 + ll;            //           col = lane%16
                const float val = acc[mi][ni][j];
                if (Cf) {
                    Cf[(size_t)m * N + n] = val + (bias ? bias[n] : 0.f);
                } else if (transC) {
                    Cb[(size_t)n * M + m] = f2bf(val);
                } else {
                    Cb[(size_t)m * N + n] = f2bf(val);
                }
            }
}

// ---------------------------------------------------------------------------
// Kernel 2: causal flash attention, one wave per (batch, head, 16-query tile).
//   Qp : [B, N, 512] bf16 (head h at column h*64)
//   Kp : [B, N, 512] bf16
//   VpT: [B, 512, N] bf16 (transposed: row = h*64+d, col = token)
//   Aout:[B, N, 512] bf16 merged-head attention output
// ---------------------------------------------------------------------------
__global__ void __launch_bounds__(32) attn_causal_kernel(
    const unsigned short* __restrict__ Qp,
    const unsigned short* __restrict__ Kp,
    const unsigned short* __restrict__ VpT,
    unsigned short* __restrict__ Aout) {
    const int Nn = 2048, C = 512, Dh = 64;
    const int b = blockIdx.z, h = blockIdx.y, qt = blockIdx.x;
    const int lane = threadIdx.x & 31;
    const int lhi = lane >> 4, ll = lane & 15;
    const int qbase = qt * 16;

    // ---- Q fragments (16 rows x 64 d = two 16x32 A-fragments) ----
    const unsigned short* Qrow = Qp + ((size_t)(b * Nn + qbase + ll)) * C + h * Dh;
    FragBF aq0, aq1;
#pragma unroll
    for (int j = 0; j < 8; ++j) {
        const int kA = 2 * j + lhi * 16;
        aq0.u[j] = *reinterpret_cast<const unsigned int*>(Qrow + kA);
        aq1.u[j] = *reinterpret_cast<const unsigned int*>(Qrow + 32 + kA);
    }

    float mrow[8], lsum[8];
    v8f o[4];
#pragma unroll
    for (int j = 0; j < 8; ++j) { mrow[j] = -1e30f; lsum[j] = 0.f; }
#pragma unroll
    for (int t = 0; t < 4; ++t)
#pragma unroll
        for (int j = 0; j < 8; ++j) o[t][j] = 0.f;

    const unsigned short* Kbase = Kp + (size_t)b * Nn * C + h * Dh;
    const unsigned short* Vbase = VpT + (size_t)b * C * Nn + (size_t)h * Dh * Nn;

    __shared__ unsigned short Pl[16 * 32];   // P tile staging (C-layout -> A-layout)

    const float SCALE = 0.125f;              // 1/sqrt(64)
    const int kcMax = (qbase + 15) >> 5;     // causal bound on 32-key chunks
    for (int kc = 0; kc <= kcMax; ++kc) {
        const int key0 = kc << 5;
        v8f s0 = {0.f, 0.f, 0.f, 0.f, 0.f, 0.f, 0.f, 0.f};
        v8f s1 = {0.f, 0.f, 0.f, 0.f, 0.f, 0.f, 0.f, 0.f};

        // ---- S = Q @ K^T over this 32-key chunk (4 WMMAs) ----
        {
            const unsigned short* Krow0 = Kbase + (size_t)(key0 + ll) * C;
            const unsigned short* Krow1 = Kbase + (size_t)(key0 + 16 + ll) * C;
            FragBF bk;
#pragma unroll
            for (int j = 0; j < 8; ++j)
                bk.u[j] = *reinterpret_cast<const unsigned int*>(Krow0 + lhi * 16 + 2 * j);
            s0 = wmma_bf16(aq0, bk, s0);
#pragma unroll
            for (int j = 0; j < 8; ++j)
                bk.u[j] = *reinterpret_cast<const unsigned int*>(Krow0 + 32 + lhi * 16 + 2 * j);
            s0 = wmma_bf16(aq1, bk, s0);
#pragma unroll
            for (int j = 0; j < 8; ++j)
                bk.u[j] = *reinterpret_cast<const unsigned int*>(Krow1 + lhi * 16 + 2 * j);
            s1 = wmma_bf16(aq0, bk, s1);
#pragma unroll
            for (int j = 0; j < 8; ++j)
                bk.u[j] = *reinterpret_cast<const unsigned int*>(Krow1 + 32 + lhi * 16 + 2 * j);
            s1 = wmma_bf16(aq1, bk, s1);
        }

        // ---- online softmax (f32), rows striped across 16-lane groups ----
        const int n0 = key0 + ll;
        const int n1 = n0 + 16;
#pragma unroll
        for (int j = 0; j < 8; ++j) {
            const int m = qbase + j + lhi * 8;
            float v0 = (n0 <= m) ? s0[j] * SCALE : -1e30f;
            float v1 = (n1 <= m) ? s1[j] * SCALE : -1e30f;
            float r = fmaxf(v0, v1);
            r = fmaxf(r, __shfl_xor(r, 1, 32));
            r = fmaxf(r, __shfl_xor(r, 2, 32));
            r = fmaxf(r, __shfl_xor(r, 4, 32));
            r = fmaxf(r, __shfl_xor(r, 8, 32));
            const float newm = fmaxf(mrow[j], r);
            const float corr = __expf(mrow[j] - newm);
            mrow[j] = newm;
            const float p0 = __expf(v0 - newm);
            const float p1 = __expf(v1 - newm);
            float rs = p0 + p1;
            rs += __shfl_xor(rs, 1, 32);
            rs += __shfl_xor(rs, 2, 32);
            rs += __shfl_xor(rs, 4, 32);
            rs += __shfl_xor(rs, 8, 32);
            lsum[j] = lsum[j] * corr + rs;
#pragma unroll
            for (int t = 0; t < 4; ++t) o[t][j] *= corr;
            const int r0 = j + lhi * 8;
            Pl[r0 * 32 + ll]      = f2bf(p0);
            Pl[r0 * 32 + 16 + ll] = f2bf(p1);
        }
        __syncthreads();

        // reload P in A-fragment layout
        FragBF ap;
#pragma unroll
        for (int j = 0; j < 8; ++j) {
            const int kA = 2 * j + lhi * 16;
            ap.u[j] = *reinterpret_cast<const unsigned int*>(&Pl[ll * 32 + kA]);
        }
        __syncthreads();

        // ---- O += P @ V (4 WMMAs, one per 16-wide d tile) ----
#pragma unroll
        for (int t = 0; t < 4; ++t) {
            const unsigned short* Vrow =
                Vbase + (size_t)(t * 16 + ll) * Nn + key0 + lhi * 16;
            FragBF bv;
#pragma unroll
            for (int j = 0; j < 8; ++j)
                bv.u[j] = *reinterpret_cast<const unsigned int*>(Vrow + 2 * j);
            o[t] = wmma_bf16(ap, bv, o[t]);
        }
    }

    // ---- normalize and store merged-head output (bf16) ----
#pragma unroll
    for (int j = 0; j < 8; ++j) {
        const float inv = 1.0f / lsum[j];
        const int m = qbase + j + lhi * 8;
        unsigned short* dst = Aout + ((size_t)(b * Nn + m)) * C + h * Dh;
#pragma unroll
        for (int t = 0; t < 4; ++t)
            dst[t * 16 + ll] = f2bf(o[t][j] * inv);
    }
}

// ---------------------------------------------------------------------------
// Host-side orchestration
// ---------------------------------------------------------------------------
extern "C" void kernel_launch(void* const* d_in, const int* in_sizes, int n_in,
                              void* d_out, int out_size, void* d_ws, size_t ws_size,
                              hipStream_t stream) {
    const float* q  = (const float*)d_in[0];
    const float* k  = (const float*)d_in[1];
    const float* v  = (const float*)d_in[2];
    const float* Wq = (const float*)d_in[3];
    const float* Wk = (const float*)d_in[4];
    const float* Wv = (const float*)d_in[5];
    const float* Wo = (const float*)d_in[6];
    const float* bo = (const float*)d_in[7];

    const int B = 2, Nn = 2048, C = 512, H = 8;
    const size_t Sx = (size_t)B * Nn * C;   // activation tensor elements (2M)
    const size_t Sw = (size_t)C * C;        // weight matrix elements (256K)

    unsigned short* ws  = (unsigned short*)d_ws;
    unsigned short* qb  = ws;                // bf16 q        (Sx)
    unsigned short* kb  = qb + Sx;           // bf16 k        (Sx)
    unsigned short* vb  = kb + Sx;           // bf16 v        (Sx)
    unsigned short* Wqb = vb + Sx;           // bf16 weights  (Sw each)
    unsigned short* Wkb = Wqb + Sw;
    unsigned short* Wvb = Wkb + Sw;
    unsigned short* Wob = Wvb + Sw;
    unsigned short* Qp  = Wob + Sw;          // Q projection  (Sx)
    unsigned short* Kp  = Qp + Sx;           // K projection  (Sx)
    unsigned short* VpT = Kp + Sx;           // V projection, transposed (Sx)
    unsigned short* Aout = qb;               // reuse qb (consumed) for attn output

    // 0) convert everything to bf16
    const int na = (int)Sx, nw = (int)Sw;
    cvt_f32_to_bf16<<<(na + 255) / 256, 256, 0, stream>>>(q, qb, na);
    cvt_f32_to_bf16<<<(na + 255) / 256, 256, 0, stream>>>(k, kb, na);
    cvt_f32_to_bf16<<<(na + 255) / 256, 256, 0, stream>>>(v, vb, na);
    cvt_f32_to_bf16<<<(nw + 255) / 256, 256, 0, stream>>>(Wq, Wqb, nw);
    cvt_f32_to_bf16<<<(nw + 255) / 256, 256, 0, stream>>>(Wk, Wkb, nw);
    cvt_f32_to_bf16<<<(nw + 255) / 256, 256, 0, stream>>>(Wv, Wvb, nw);
    cvt_f32_to_bf16<<<(nw + 255) / 256, 256, 0, stream>>>(Wo, Wob, nw);

    // 1) QKV projections (x @ W.T), per batch
    const dim3 gg(Nn / 64, C / 64), gb(128);
    for (int b = 0; b < B; ++b) {
        const size_t ao = (size_t)b * Nn * C;
        gemm_bf16_wmma<<<gg, gb, 0, stream>>>(qb + ao, Wqb, Qp + ao, nullptr,
                                              nullptr, Nn, C, C, 0);
        gemm_bf16_wmma<<<gg, gb, 0, stream>>>(kb + ao, Wkb, Kp + ao, nullptr,
                                              nullptr, Nn, C, C, 0);
        gemm_bf16_wmma<<<gg, gb, 0, stream>>>(vb + ao, Wvb, VpT + (size_t)b * C * Nn,
                                              nullptr, nullptr, Nn, C, C, /*transC=*/1);
    }

    // 2) causal flash attention
    attn_causal_kernel<<<dim3(Nn / 16, H, B), 32, 0, stream>>>(Qp, Kp, VpT, Aout);

    // 3) output projection + bias, f32 output
    for (int b = 0; b < B; ++b) {
        const size_t ao = (size_t)b * Nn * C;
        gemm_bf16_wmma<<<gg, gb, 0, stream>>>(Aout + ao, Wob, nullptr,
                                              (float*)d_out + ao, bo, Nn, C, C, 0);
    }
}
